// HypergraphAttention_31533649887507
// MI455X (gfx1250) — matrix operand order repaired and checked by reading, not compiled, not run on hardware
//
#include <hip/hip_runtime.h>

typedef __attribute__((ext_vector_type(16))) _Float16 v16h;
typedef __attribute__((ext_vector_type(8)))  _Float16 v8h;
typedef __attribute__((ext_vector_type(8)))  float    v8f;
typedef __attribute__((ext_vector_type(4)))  float    v4f;
typedef __attribute__((ext_vector_type(4)))  int      v4i;

#define N_SAMPLES 8192
#define N_EDGES   4096
#define IN_DIM    512
#define DK_DIM    64
#define OUT_DIM   256
#define N_WAVES   8                      // sample-split waves per edge tile
#define S_SLICE   (N_SAMPLES / N_WAVES)  // 1024 samples per wave

__device__ __forceinline__ float lane_max16(float v) {
#pragma unroll
  for (int off = 8; off >= 1; off >>= 1)
    v = fmaxf(v, __shfl_xor(v, off, 16));
  return v;
}
__device__ __forceinline__ float lane_sum16(float v) {
#pragma unroll
  for (int off = 8; off >= 1; off >>= 1)
    v += __shfl_xor(v, off, 16);
  return v;
}

// ---------------------------------------------------------------------------
// W (K x N, f32, row-major)  ->  Wt (N x K, f16, row-major)
// ---------------------------------------------------------------------------
__global__ __launch_bounds__(256) void wt_convert_kernel(
    const float* __restrict__ W, _Float16* __restrict__ Wt, int K, int N) {
  int idx = blockIdx.x * blockDim.x + threadIdx.x;
  if (idx >= K * N) return;
  int n = idx / K;
  int k = idx - n * K;
  Wt[idx] = (_Float16)W[(size_t)k * N + n];
}

// ---------------------------------------------------------------------------
// Projection: Out = A(M x 512, f32) @ W(512 x N) using Wt (N x 512, f16).
// One wave computes one 16x16 output tile via 16 chained WMMAs (K=32).
// transpose_out=0: Out[m][n] (ld=ldo); transpose_out=1: Out[n][m] (ld=ldo).
// ---------------------------------------------------------------------------
__global__ __launch_bounds__(32) void proj16_kernel(
    const float* __restrict__ A, const _Float16* __restrict__ Wt,
    _Float16* __restrict__ Out, int ldo, int transpose_out) {
  const int l  = threadIdx.x;      // 0..31
  const int hi = (l >= 16) ? 1 : 0;
  const int ln = l & 15;
  const int m0 = blockIdx.x * 16;
  const int n0 = blockIdx.y * 16;

  const float*    arow = A  + (size_t)(m0 + ln) * IN_DIM;
  const _Float16* brow = Wt + (size_t)(n0 + ln) * IN_DIM;

  v8f acc = {};
#pragma unroll 4
  for (int k0 = 0; k0 < IN_DIM; k0 += 32) {
    // A fragment: lane ln = row m, K chunks {ka..ka+7, ka+16..ka+23}
    const int ka = k0 + (hi ? 8 : 0);
    v4f a0 = *(const v4f*)(arow + ka);
    v4f a1 = *(const v4f*)(arow + ka + 4);
    v4f a2 = *(const v4f*)(arow + ka + 16);
    v4f a3 = *(const v4f*)(arow + ka + 20);
    v16h af;
#pragma unroll
    for (int i = 0; i < 4; ++i) {
      af[i]      = (_Float16)a0[i];
      af[4 + i]  = (_Float16)a1[i];
      af[8 + i]  = (_Float16)a2[i];
      af[12 + i] = (_Float16)a3[i];
    }
    // B fragment: lane ln = column n, K chunks {kb..kb+7, kb+16..kb+23} from Wt row
    const int kb = k0 + (hi ? 8 : 0);
    v8h b0 = *(const v8h*)(brow + kb);
    v8h b1 = *(const v8h*)(brow + kb + 16);
    v16h bf;
#pragma unroll
    for (int i = 0; i < 8; ++i) { bf[i] = b0[i]; bf[8 + i] = b1[i]; }

    acc = __builtin_amdgcn_wmma_f32_16x16x32_f16(
        false, af, false, bf, (short)0, acc, false, false);
  }

#pragma unroll
  for (int r = 0; r < 8; ++r) {
    const int m = m0 + r + hi * 8;
    const int n = n0 + ln;
    if (transpose_out)
      Out[(size_t)n * ldo + m] = (_Float16)acc[r];
    else
      Out[(size_t)m * ldo + n] = (_Float16)acc[r];
  }
}

// ---------------------------------------------------------------------------
// Masked flash attention, split-K over samples (flash-decoding style).
// Block = 8 waves, one 16-edge tile. Wave w streams samples
// [w*1024, (w+1)*1024) with online softmax + WMMA, keeping private
// (m, l, acc[16 edges x 256 dims]). Final merge in LDS:
//   M = max_w m_w ; L = sum_w l_w e^{m_w-M} ; out = sum_w acc_w e^{m_w-M} / L
// ---------------------------------------------------------------------------
__global__ __launch_bounds__(256) void hyperattn_kernel(
    const int* __restrict__ H, const _Float16* __restrict__ Qh,
    const _Float16* __restrict__ Kh, const _Float16* __restrict__ Vt,
    float* __restrict__ Out) {
  __shared__ __align__(16) _Float16 Plds[N_WAVES][16 * 32];  // P staging
  __shared__ float m_st[N_WAVES][16];                        // per-wave row max
  __shared__ float l_st[N_WAVES][16];                        // per-wave row sum
  __shared__ float accbuf[16 * OUT_DIM];                     // merged 16x256

  const int tid = threadIdx.x;
  const int w   = tid >> 5;   // wave id: 0..7
  const int l   = tid & 31;   // lane
  const int hi  = (l >= 16) ? 1 : 0;
  const int ln  = l & 15;
  const int e0  = blockIdx.x * 16;
  const float NEG_INF = -__builtin_inff();
  const float scale = 0.125f;  // 1/sqrt(64)

  // zero the merge buffer (16 entries per thread)
#pragma unroll
  for (int i = 0; i < 16; ++i) accbuf[tid * 16 + i] = 0.f;

  // Q fragments, resident for the whole loop (dk=64 -> two K=32 chunks)
  v16h qf[2];
  {
    const _Float16* qrow = Qh + (size_t)(e0 + ln) * DK_DIM;
#pragma unroll
    for (int c = 0; c < 2; ++c) {
      const int ka = c * 32 + (hi ? 8 : 0);
      v8h q0 = *(const v8h*)(qrow + ka);
      v8h q1 = *(const v8h*)(qrow + ka + 16);
#pragma unroll
      for (int i = 0; i < 8; ++i) { qf[c][i] = q0[i]; qf[c][8 + i] = q1[i]; }
    }
  }

  float m_i[8], l_i[8];
  v8f acc[16];
#pragma unroll
  for (int r = 0; r < 8; ++r) { m_i[r] = NEG_INF; l_i[r] = 0.f; }
#pragma unroll
  for (int j = 0; j < 16; ++j) acc[j] = (v8f){};

  __syncthreads();  // accbuf zeroed before any merge atomics

  const int s_lo = w * S_SLICE;
  const int s_hi = s_lo + S_SLICE;
  for (int s0 = s_lo; s0 < s_hi; s0 += 32) {
    // Prefetch next chunk's H rows (one row per lane -> 32 rows)
    if (s0 + 32 < s_hi)
      __builtin_prefetch(H + (size_t)(s0 + 32 + l) * N_EDGES + e0, 0, 0);

    // ---- masked scores for two 16-sample subtiles ----
    float st[2][8];
#pragma unroll
    for (int sub = 0; sub < 2; ++sub) {
      const int sbase = s0 + sub * 16;
      v8f s = {};
      const _Float16* krow = Kh + (size_t)(sbase + ln) * DK_DIM;
#pragma unroll
      for (int c = 0; c < 2; ++c) {
        const int kb = c * 32 + (hi ? 8 : 0);
        v8h b0 = *(const v8h*)(krow + kb);
        v8h b1 = *(const v8h*)(krow + kb + 16);
        v16h bk;
#pragma unroll
        for (int i = 0; i < 8; ++i) { bk[i] = b0[i]; bk[8 + i] = b1[i]; }
        s = __builtin_amdgcn_wmma_f32_16x16x32_f16(
            false, qf[c], false, bk, (short)0, s, false, false);
      }
      // mask: lane holds column (sample sbase+ln), rows e0+hi*8+r
      const int* hrow = H + (size_t)(sbase + ln) * N_EDGES + e0 + hi * 8;
      v4i h0 = __builtin_nontemporal_load((const v4i*)hrow);
      v4i h1 = __builtin_nontemporal_load((const v4i*)(hrow + 4));
#pragma unroll
      for (int r = 0; r < 4; ++r) {
        st[sub][r]     = h0[r] ? s[r] * scale     : NEG_INF;
        st[sub][4 + r] = h1[r] ? s[4 + r] * scale : NEG_INF;
      }
    }

    // ---- online softmax update (rows = r + hi*8, reduce over 16 lanes) ----
    float alpha[8];
#pragma unroll
    for (int r = 0; r < 8; ++r) {
      float rmax = lane_max16(fmaxf(st[0][r], st[1][r]));
      float mnew = fmaxf(m_i[r], rmax);
      alpha[r] = (mnew == NEG_INF) ? 1.f : __expf(m_i[r] - mnew);
      m_i[r] = mnew;
    }
    float psum[8];
#pragma unroll
    for (int r = 0; r < 8; ++r) psum[r] = 0.f;
#pragma unroll
    for (int sub = 0; sub < 2; ++sub) {
#pragma unroll
      for (int r = 0; r < 8; ++r) {
        float pv = (m_i[r] == NEG_INF) ? 0.f : __expf(st[sub][r] - m_i[r]);
        psum[r] += pv;
        Plds[w][(r + hi * 8) * 32 + sub * 16 + ln] = (_Float16)pv;
      }
    }
#pragma unroll
    for (int r = 0; r < 8; ++r)
      l_i[r] = l_i[r] * alpha[r] + lane_sum16(psum[r]);
    // rescale running accumulator
#pragma unroll
    for (int j = 0; j < 16; ++j)
#pragma unroll
      for (int r = 0; r < 8; ++r) acc[j][r] *= alpha[r];

    // Wave-local LDS store->load ordering: DS ops are in-order per wave;
    // explicit split-counter wait (no cross-wave barrier needed).
    __asm__ volatile("s_wait_dscnt 0" ::: "memory");

    // P tile as A-fragment (16 edges x 32 samples)
    v16h pa;
    {
      const int koff = hi ? 8 : 0;
      v8h p0 = *(const v8h*)&Plds[w][ln * 32 + koff];
      v8h p1 = *(const v8h*)&Plds[w][ln * 32 + koff + 16];
#pragma unroll
      for (int i = 0; i < 8; ++i) { pa[i] = p0[i]; pa[8 + i] = p1[i]; }
    }

    // ---- acc += P @ V  (B from Vt: K chunks {8h..8h+7, 8h+16..8h+23} per lane) ----
#pragma unroll
    for (int jt = 0; jt < 16; ++jt) {
      const _Float16* vrow =
          Vt + (size_t)(jt * 16 + ln) * N_SAMPLES + s0 + (hi ? 8 : 0);
      v8h b0 = *(const v8h*)vrow;
      v8h b1 = *(const v8h*)(vrow + 16);
      v16h bv;
#pragma unroll
      for (int i = 0; i < 8; ++i) { bv[i] = b0[i]; bv[8 + i] = b1[i]; }
      acc[jt] = __builtin_amdgcn_wmma_f32_16x16x32_f16(
          false, pa, false, bv, (short)0, acc[jt], false, false);
    }
  }

  // ---- publish per-wave softmax stats ----
  if (ln == 0) {
#pragma unroll
    for (int r = 0; r < 8; ++r) {
      m_st[w][r + hi * 8] = m_i[r];
      l_st[w][r + hi * 8] = l_i[r];
    }
  }
  __syncthreads();

  // ---- merge: rescale private acc by e^{m_w - M}, LDS-atomic accumulate ----
  float fw[8];
#pragma unroll
  for (int r = 0; r < 8; ++r) {
    const int row = r + hi * 8;
    float M = NEG_INF;
#pragma unroll
    for (int ww = 0; ww < N_WAVES; ++ww) M = fmaxf(M, m_st[ww][row]);
    fw[r] = (M == NEG_INF) ? 0.f : __expf(m_i[r] - M);
  }
#pragma unroll
  for (int jt = 0; jt < 16; ++jt) {
#pragma unroll
    for (int r = 0; r < 8; ++r) {
      const int row = r + hi * 8;
      atomicAdd(&accbuf[row * OUT_DIM + jt * 16 + ln], acc[jt][r] * fw[r]);
    }
  }
  __syncthreads();

  // ---- normalize and store (streaming NT stores) ----
  {
    const int row  = tid >> 4;         // 0..15
    const int cb   = (tid & 15) * 16;  // column base
    float M = NEG_INF, L = 0.f;
#pragma unroll
    for (int ww = 0; ww < N_WAVES; ++ww) M = fmaxf(M, m_st[ww][row]);
#pragma unroll
    for (int ww = 0; ww < N_WAVES; ++ww)
      L += (M == NEG_INF) ? 0.f : l_st[ww][row] * __expf(m_st[ww][row] - M);
    const float inv = 1.0f / L;
    float* orow = Out + (size_t)(e0 + row) * OUT_DIM + cb;
#pragma unroll
    for (int i = 0; i < 16; ++i)
      __builtin_nontemporal_store(accbuf[row * OUT_DIM + cb + i] * inv,
                                  orow + i);
  }
}

// ---------------------------------------------------------------------------
extern "C" void kernel_launch(void* const* d_in, const int* in_sizes, int n_in,
                              void* d_out, int out_size, void* d_ws,
                              size_t ws_size, hipStream_t stream) {
  const float* X  = (const float*)d_in[0];  // 8192 x 512
  const float* Y  = (const float*)d_in[1];  // 4096 x 512
  const int*   H  = (const int*)d_in[2];    // 8192 x 4096
  const float* Wq = (const float*)d_in[3];  // 512 x 64
  const float* Wk = (const float*)d_in[4];  // 512 x 64
  const float* Wv = (const float*)d_in[5];  // 512 x 256
  float* Out = (float*)d_out;               // 4096 x 256

  _Float16* ws  = (_Float16*)d_ws;
  _Float16* Wqt = ws;                        //  64 x 512
  _Float16* Wkt = Wqt + 64 * IN_DIM;         //  64 x 512
  _Float16* Wvt = Wkt + 64 * IN_DIM;         // 256 x 512
  _Float16* Qh  = Wvt + OUT_DIM * IN_DIM;    // 4096 x 64
  _Float16* Kh  = Qh + (size_t)N_EDGES * DK_DIM;    // 8192 x 64
  _Float16* Vt  = Kh + (size_t)N_SAMPLES * DK_DIM;  // 256 x 8192

  // Stage 0: weight transpose+convert (tiny)
  wt_convert_kernel<<<(64 * IN_DIM + 255) / 256, 256, 0, stream>>>(Wq, Wqt, IN_DIM, 64);
  wt_convert_kernel<<<(64 * IN_DIM + 255) / 256, 256, 0, stream>>>(Wk, Wkt, IN_DIM, 64);
  wt_convert_kernel<<<(OUT_DIM * IN_DIM + 255) / 256, 256, 0, stream>>>(Wv, Wvt, IN_DIM, OUT_DIM);

  // Stage 1: projections (WMMA)
  dim3 gq(N_EDGES / 16, DK_DIM / 16);
  proj16_kernel<<<gq, 32, 0, stream>>>(Y, Wqt, Qh, DK_DIM, 0);
  dim3 gk(N_SAMPLES / 16, DK_DIM / 16);
  proj16_kernel<<<gk, 32, 0, stream>>>(X, Wkt, Kh, DK_DIM, 0);
  dim3 gv(N_SAMPLES / 16, OUT_DIM / 16);
  proj16_kernel<<<gv, 32, 0, stream>>>(X, Wvt, Vt, N_SAMPLES, 1);  // Vt = (X Wv)^T

  // Stage 2: masked flash attention, 8-way sample split per edge tile
  hyperattn_kernel<<<N_EDGES / 16, 32 * N_WAVES, 0, stream>>>(H, Qh, Kh, Vt, Out);
}
